// UVIT_1803886265727
// MI455X (gfx1250) — matrix operand order repaired and checked by reading, not compiled, not run on hardware
//
#include <hip/hip_runtime.h>
#include <hip/hip_bf16.h>
#include <math.h>

// ---------------------------------------------------------------------------
// CDNA5 (gfx1250) fused adaLN feed-forward, f16 WMMA with f32 accumulation.
// ---------------------------------------------------------------------------

typedef __attribute__((ext_vector_type(16))) _Float16 v16h;
typedef __attribute__((ext_vector_type(8)))  _Float16 v8h;
typedef __attribute__((ext_vector_type(4)))  _Float16 v4h;
typedef __attribute__((ext_vector_type(8)))  float    v8f;
typedef __attribute__((ext_vector_type(4)))  float    v4f;

#define D_MODEL 512
#define HID     2048
#define COND    2048
#define N_TOK   40000     // tokens per batch element
#define MT      16        // row-tile per workgroup
#define XN_S    520       // padded f16 stride (1040B = 16*65 -> aligned, conflict-free)
#define H_S     2056      // padded f16 stride (4112B = 16*257)

__device__ __forceinline__ float silu_f(float v){ return v / (1.0f + __expf(-v)); }

// ---- f32 -> f16 cast ------------------------------------------------------
__global__ __launch_bounds__(256) void k_cvt(const float* __restrict__ s,
                                             _Float16* __restrict__ d, int n){
  int i = blockIdx.x * 256 + threadIdx.x;
  if (i < n) d[i] = (_Float16)s[i];
}

// ---- silu(t) --------------------------------------------------------------
__global__ __launch_bounds__(256) void k_silu(const float* __restrict__ s,
                                              float* __restrict__ d, int n){
  int i = blockIdx.x * 256 + threadIdx.x;
  if (i < n) d[i] = silu_f(s[i]);
}

// ---- ss[layer][b][4096] = silu(t[b]) . w_ss[layer][row] + b_ss[layer][row]
// one wave per output element, 16384 outputs total
__global__ __launch_bounds__(256) void k_ss(const float* __restrict__ st,
                                            const float* __restrict__ wss,
                                            const float* __restrict__ bss,
                                            float* __restrict__ ss){
  int o     = blockIdx.x * 8 + (threadIdx.x >> 5);
  int lane  = threadIdx.x & 31;
  int layer = o >> 13;
  int b     = (o >> 12) & 1;
  int row   = o & 4095;
  const float* w = wss + ((size_t)layer * 2 * HID + row) * COND;
  const float* s = st + b * COND;
  float acc = 0.f;
#pragma unroll
  for (int c = 0; c < 16; ++c){
    int idx = c * 128 + lane * 4;
    v4f a  = *(const v4f*)(w + idx);
    v4f tt = *(const v4f*)(s + idx);
    acc += a.x * tt.x + a.y * tt.y + a.z * tt.z + a.w * tt.w;
  }
#pragma unroll
  for (int off = 16; off; off >>= 1) acc += __shfl_xor(acc, off, 32);
  if (lane == 0) ss[o] = acc + bss[layer * 2 * HID + row];
}

// ---- fused FF block: out = (silu(rms(x)@Wi^T)*(1+sc)+sh) @ Wo^T + x -------
// Wi: f16 [HID][D_MODEL] row-major; Wo: f16 [D_MODEL][HID] row-major.
// ssL: f32 [2][2*HID] for this layer.  grid = 80000/MT blocks of 256.
__global__ __launch_bounds__(256) void k_ff(const float* __restrict__ xin,
                                            float* __restrict__ xout,
                                            const _Float16* __restrict__ wi,
                                            const _Float16* __restrict__ wo,
                                            const float* __restrict__ ssL){
  __shared__ _Float16 xnS[MT * XN_S];   // rmsnormed activations, f16
  __shared__ _Float16 hS [MT * H_S];    // hidden activations, f16

  const int tid  = threadIdx.x;
  const int wv   = tid >> 5;
  const int lane = tid & 31;
  const int hf   = lane >> 4;          // lane half (K-half selector)
  const int l16  = lane & 15;
  const long tile0 = (long)blockIdx.x * MT;
  const int b = (tile0 >= N_TOK) ? 1 : 0;
  const float* ssb = ssL + b * 2 * HID;

  // ---- Phase 1: RMSNorm (no scale) -> f16 LDS.  8 waves x 2 rows. --------
#pragma unroll
  for (int rr = 0; rr < 2; ++rr){
    int row = wv * 2 + rr;
    const float* xr = xin + (tile0 + row) * D_MODEL;
    v4f v[4];
    float s = 0.f;
#pragma unroll
    for (int c = 0; c < 4; ++c){
      v[c] = *(const v4f*)(xr + c * 128 + lane * 4);
      s += v[c].x * v[c].x + v[c].y * v[c].y + v[c].z * v[c].z + v[c].w * v[c].w;
    }
#pragma unroll
    for (int off = 16; off; off >>= 1) s += __shfl_xor(s, off, 32);
    float r = rsqrtf(s * (1.0f / D_MODEL) + 1e-12f);
    _Float16* dst = xnS + row * XN_S;
#pragma unroll
    for (int c = 0; c < 4; ++c){
      v4h o;
      o.x = (_Float16)(v[c].x * r); o.y = (_Float16)(v[c].y * r);
      o.z = (_Float16)(v[c].z * r); o.w = (_Float16)(v[c].w * r);
      *(v4h*)(dst + c * 128 + lane * 4) = o;
    }
  }
  __syncthreads();

  // ---- Phase 2: H = silu(xn @ Wi^T)*(1+scale)+shift   M=16 N=2048 K=512 --
  {
    const _Float16* aB = xnS + l16 * XN_S;
    for (int it = 0; it < 4; ++it){
      int n0 = (wv + it * 8) * 64;           // 64-column quad per wave
      v8f acc[4] = {};
      for (int k = 0; k < D_MODEL; k += 32){
        union { v16h v; v8h h[2]; } au;
        au.h[0] = *(const v8h*)(aB + k + hf * 16);        // K 16h+0..7
        au.h[1] = *(const v8h*)(aB + k + hf * 16 + 8);    // K 16h+8..15
        const _Float16* bp = wi + (size_t)(n0 + l16) * D_MODEL + k + hf * 16;
#pragma unroll
        for (int j = 0; j < 4; ++j){
          v16h bf = *(const v16h*)(bp + (size_t)j * 16 * D_MODEL);
          acc[j] = __builtin_amdgcn_wmma_f32_16x16x32_f16(
              false, au.v, false, bf, (short)0, acc[j], false, false);
        }
      }
#pragma unroll
      for (int j = 0; j < 4; ++j){
        int col  = n0 + j * 16 + l16;
        float sc = ssb[col] + 1.0f;
        float sh = ssb[HID + col];
#pragma unroll
        for (int rI = 0; rI < 8; ++rI){
          float hv = silu_f(acc[j][rI]) * sc + sh;
          hS[(rI + 8 * hf) * H_S + col] = (_Float16)hv;
        }
      }
    }
  }
  __syncthreads();

  // ---- Phase 3: out = H @ Wo^T + x     M=16 N=512 K=2048 -----------------
  {
    const _Float16* aB = hS + l16 * H_S;
    int n0 = wv * 64;                        // 8 waves cover 512 columns
    v8f acc[4] = {};
    for (int k = 0; k < HID; k += 32){
      union { v16h v; v8h h[2]; } au;
      au.h[0] = *(const v8h*)(aB + k + hf * 16);
      au.h[1] = *(const v8h*)(aB + k + hf * 16 + 8);
      const _Float16* bp = wo + (size_t)(n0 + l16) * HID + k + hf * 16;
#pragma unroll
      for (int j = 0; j < 4; ++j){
        v16h bf = *(const v16h*)(bp + (size_t)j * 16 * HID);
        acc[j] = __builtin_amdgcn_wmma_f32_16x16x32_f16(
            false, au.v, false, bf, (short)0, acc[j], false, false);
      }
    }
#pragma unroll
    for (int j = 0; j < 4; ++j){
      int col = n0 + j * 16 + l16;
#pragma unroll
      for (int rI = 0; rI < 8; ++rI){
        long grow = tile0 + rI + 8 * hf;
        float res = xin[grow * D_MODEL + col];
        xout[grow * D_MODEL + col] = acc[j][rI] + res;
      }
    }
  }
}

// ---------------------------------------------------------------------------
extern "C" void kernel_launch(void* const* d_in, const int* in_sizes, int n_in,
                              void* d_out, int out_size, void* d_ws, size_t ws_size,
                              hipStream_t stream) {
  const float* x     = (const float*)d_in[0];   // [2,40000,512]
  const float* t     = (const float*)d_in[1];   // [2,2048]
  const float* w_in  = (const float*)d_in[2];   // [2,2048,512]
  const float* w_out = (const float*)d_in[3];   // [2,512,2048]
  const float* w_ss  = (const float*)d_in[4];   // [2,4096,2048]
  const float* b_ss  = (const float*)d_in[5];   // [2,4096]
  float* out = (float*)d_out;                   // [2,40000,512]
  (void)in_sizes; (void)n_in; (void)out_size; (void)ws_size;

  char* ws = (char*)d_ws;
  _Float16* wi16 = (_Float16*)(ws);                           // 2*2048*512  f16 (4 MiB)
  _Float16* wo16 = (_Float16*)(ws + (4u << 20));              // 2*512*2048  f16 (4 MiB)
  float*    st   = (float*)   (ws + (8u << 20));              // 2*2048      f32
  float*    ss   = (float*)   (ws + (8u << 20) + 16384);      // 2*2*4096    f32

  const int nW = 2 * HID * D_MODEL;   // 2,097,152 per weight tensor
  k_cvt <<<(nW + 255) / 256, 256, 0, stream>>>(w_in,  wi16, nW);
  k_cvt <<<(nW + 255) / 256, 256, 0, stream>>>(w_out, wo16, nW);
  k_silu<<<(2 * COND + 255) / 256, 256, 0, stream>>>(t, st, 2 * COND);
  k_ss  <<<2048, 256, 0, stream>>>(st, w_ss, b_ss, ss);       // 16384 outputs, 8 waves/blk

  const int nTiles = (2 * N_TOK) / MT;  // 5000
  // layer 0: x -> out
  k_ff<<<nTiles, 256, 0, stream>>>(x,   out, wi16,                 wo16,                 ss);
  // layer 1: out -> out (in-place safe: each WG touches only its own rows)
  k_ff<<<nTiles, 256, 0, stream>>>(out, out, wi16 + (size_t)HID * D_MODEL,
                                   wo16 + (size_t)D_MODEL * HID,  ss + 2 * 2 * HID);
}
